// GCN_40690520162672
// MI455X (gfx1250) — matrix operand order repaired and checked by reading, not compiled, not run on hardware
//
#include <hip/hip_runtime.h>

// ---------------------------------------------------------------------------
// GCN forward on MI455X (gfx1250, wave32) via V_WMMA_F32_16X16X32_F16.
//   T1  = x @ W1                         [10000,512]x[512,256]
//   H   = relu(adj @ T1 + b1)            [10000,10000]x[10000,256]
//   T2  = H @ W2                         [10000,256]x[256,64]
//   out = adj @ T2 + b2                  [10000,10000]x[10000,64]
// One parameterized f16-WMMA GEMM (f32 in/out, f32 accumulate), software-
// pipelined with LDS double-buffering; in-register B transpose; 64-row
// M-tiles (2x4 wave grid, 2x NT WMMAs per wave per K-step).
// ---------------------------------------------------------------------------

typedef __attribute__((ext_vector_type(16))) _Float16 v16h;
typedef __attribute__((ext_vector_type(8)))  _Float16 v8h;
typedef __attribute__((ext_vector_type(4)))  _Float16 v4h;
typedef __attribute__((ext_vector_type(2)))  _Float16 v2h;
typedef __attribute__((ext_vector_type(8)))  float    v8f;
typedef __attribute__((ext_vector_type(4)))  float    f32x4;

#define MTILE 64
#define KTILE 32
#define LROW  40   // LDS row pitch in halfs: 32 data + 8 pad = 80B (16B aligned)

// One K-step: load all fragments, then MT x NT WMMAs (loads batched so the
// scheduler can use partial dscnt waits and back-to-back WMMAs).
template <int C, int MT, int NT>
__device__ __forceinline__
void wmma_step(const _Float16 (*__restrict__ As)[LROW],
               const _Float16 (*__restrict__ Bs)[LROW],
               int wm, int wn, int l15, int half, v8f (&acc)[MT][NT])
{
    // A fragments (ISA 16-bit A 16x32 layout):
    // lanes 0-15: row=lane, K 0-7 & 16-23 ; lanes 16-31: row=lane-16, K 8-15 & 24-31
    v16h af[MT];
#pragma unroll
    for (int mt = 0; mt < MT; ++mt) {
        const char* ab =
            (const char*)&As[wm * 32 + mt * 16 + l15][0] + half * 16;
        v8h a0 = *(const v8h*)(ab);          // ds_load_b128
        v8h a1 = *(const v8h*)(ab + 32);     // ds_load_b128
        af[mt] = __builtin_shufflevector(a0, a1,
                  0,1,2,3,4,5,6,7,8,9,10,11,12,13,14,15);
    }
    // B fragments (32x16): lanes 0-15 col=lane K 0-7 & 16-23; lanes 16-31 col K 8-15 & 24-31
    v16h bf[NT];
#pragma unroll
    for (int t = 0; t < NT; ++t) {
        const int n = wn * (C / 4) + t * 16 + l15;
        const char* bb = (const char*)&Bs[n][0] + half * 16;
        v8h b0 = *(const v8h*)(bb);
        v8h b1 = *(const v8h*)(bb + 32);
        bf[t] = __builtin_shufflevector(b0, b1,
                 0,1,2,3,4,5,6,7,8,9,10,11,12,13,14,15);
    }
#pragma unroll
    for (int mt = 0; mt < MT; ++mt)
#pragma unroll
        for (int t = 0; t < NT; ++t)
            acc[mt][t] = __builtin_amdgcn_wmma_f32_16x16x32_f16(
                false, af[mt], false, bf[t], (short)0, acc[mt][t], false, false);
}

// D[M,C] = A[M,K] (row-major, lda=K) * B[K,C] (row-major) [+ bias][ReLU]
// 256 threads = 8 waves as 2(M) x 4(N); block tile 64 x C (full width -> adj
// streams from HBM exactly once per aggregation GEMM).
template <int C, bool RELU>
__global__ __launch_bounds__(256)
void gemm_wmma_f16(const float* __restrict__ A, const float* __restrict__ B,
                   const float* __restrict__ bias, float* __restrict__ D,
                   int M, int K)
{
    __shared__ _Float16 As[2][MTILE][LROW];
    __shared__ _Float16 Bs[2][C][LROW];

    const int tid  = threadIdx.x;
    const int lane = tid & 31;
    const int wave = tid >> 5;
    const int wm   = wave >> 2;           // 0..1  (M)
    const int wn   = wave & 3;            // 0..3  (N)
    const int l15  = lane & 15;
    const int half = lane >> 4;
    constexpr int MT = 2;                 // 16-row M tiles per wave
    constexpr int NT = C / 64;            // 16-col N tiles per wave

    // B staging geometry: each thread owns a KPT(K) x 4(N) sub-block.
    constexpr int NGRP = C / 4;           // column groups of 4
    constexpr int KGRP = 256 / NGRP;      // k groups
    constexpr int KPT  = KTILE / KGRP;    // k's per thread: C=256 -> 8, C=64 -> 2

    const int mBase = blockIdx.x * MTILE;

    // A staging map: thread -> (rows ar, ar+32; col group of 4)
    const int ar = tid >> 3;
    const int ac = (tid & 7) << 2;
    // Clamp A rows: edge-block OOB rows read valid (wrong) data whose results
    // are never stored -> no guards in the hot loop.
    const int gm0 = (mBase + ar      < M) ? (mBase + ar)      : (M - 1);
    const int gm1 = (mBase + 32 + ar < M) ? (mBase + 32 + ar) : (M - 1);
    const float* __restrict__ aRow0 = A + (size_t)gm0 * (size_t)K;
    const float* __restrict__ aRow1 = A + (size_t)gm1 * (size_t)K;

    // B staging map: thread -> (n = bn..bn+3, k = bk..bk+KPT-1)
    const int bn = (tid % NGRP) * 4;
    const int bk = (tid / NGRP) * KPT;
    const float* __restrict__ bBase = B + (size_t)bk * C + bn;

    v8f acc[MT][NT];
#pragma unroll
    for (int mt = 0; mt < MT; ++mt)
#pragma unroll
        for (int t = 0; t < NT; ++t) acc[mt][t] = v8f{};

    const int nFull = K / KTILE;          // >= 8 for all our shapes
    const int kTail = K - nFull * KTILE;  // 0 or 16

    f32x4 apre0, apre1;
    f32x4 bpre[KPT];

    // ---- preload tile 0 ----
    apre0 = *(const f32x4*)(aRow0 + ac);
    apre1 = *(const f32x4*)(aRow1 + ac);
#pragma unroll
    for (int i = 0; i < KPT; ++i)
        bpre[i] = *(const f32x4*)(bBase + (size_t)i * C);

    // ---- pipelined main loop: one barrier per K-step ----
    for (int it = 0; it < nFull; ++it) {
        const int cur = it & 1;

        // commit A tile (f32 -> f16): two ds_store_b64
        {
            v4h h0, h1;
            h0.x = (_Float16)apre0.x; h0.y = (_Float16)apre0.y;
            h0.z = (_Float16)apre0.z; h0.w = (_Float16)apre0.w;
            h1.x = (_Float16)apre1.x; h1.y = (_Float16)apre1.y;
            h1.z = (_Float16)apre1.z; h1.w = (_Float16)apre1.w;
            *(v4h*)&As[cur][ar][ac]      = h0;
            *(v4h*)&As[cur][ar + 32][ac] = h1;
        }
        // commit B tile transposed: in-register K-major packing, 4 wide stores
#pragma unroll
        for (int c = 0; c < 4; ++c) {
            if constexpr (KPT == 8) {
                v8h h;
#pragma unroll
                for (int i = 0; i < 8; ++i) h[i] = (_Float16)bpre[i][c];
                *(v8h*)&Bs[cur][bn + c][bk] = h;     // ds_store_b128
            } else {
                v2h h;
                h[0] = (_Float16)bpre[0][c];
                h[1] = (_Float16)bpre[1][c];
                *(v2h*)&Bs[cur][bn + c][bk] = h;     // ds_store_b32
            }
        }
        __syncthreads();

        // issue next tile's global loads; waited on only at next commit,
        // so HBM latency overlaps the WMMA work below
        if (it + 1 < nFull) {
            const int k0 = (it + 1) * KTILE;
            apre0 = *(const f32x4*)(aRow0 + k0 + ac);
            apre1 = *(const f32x4*)(aRow1 + k0 + ac);
#pragma unroll
            for (int i = 0; i < KPT; ++i)
                bpre[i] = *(const f32x4*)(bBase + (size_t)(k0 + i) * C);
        }

        wmma_step<C, MT, NT>(As[cur], Bs[cur], wm, wn, l15, half, acc);
    }

    // ---- guarded K tail (runs at most once; only K=10000 -> tail of 16) ----
    if (kTail) {
        const int cur = nFull & 1;
        const int k0  = nFull * KTILE;
        {
            f32x4 v0 = {0.f, 0.f, 0.f, 0.f};
            f32x4 v1 = {0.f, 0.f, 0.f, 0.f};
            const int gk = k0 + ac;
            if (gk + 3 < K) {
                v0 = *(const f32x4*)(aRow0 + gk);
                v1 = *(const f32x4*)(aRow1 + gk);
            }
            v4h h0, h1;
            h0.x = (_Float16)v0.x; h0.y = (_Float16)v0.y;
            h0.z = (_Float16)v0.z; h0.w = (_Float16)v0.w;
            h1.x = (_Float16)v1.x; h1.y = (_Float16)v1.y;
            h1.z = (_Float16)v1.z; h1.w = (_Float16)v1.w;
            *(v4h*)&As[cur][ar][ac]      = h0;
            *(v4h*)&As[cur][ar + 32][ac] = h1;
        }
#pragma unroll
        for (int i = 0; i < KPT; ++i) {
            f32x4 z = {0.f, 0.f, 0.f, 0.f};
            bpre[i] = (k0 + bk + i < K)
                    ? *(const f32x4*)(bBase + (size_t)(k0 + i) * C) : z;
        }
#pragma unroll
        for (int c = 0; c < 4; ++c) {
            if constexpr (KPT == 8) {
                v8h h;
#pragma unroll
                for (int i = 0; i < 8; ++i) h[i] = (_Float16)bpre[i][c];
                *(v8h*)&Bs[cur][bn + c][bk] = h;
            } else {
                v2h h;
                h[0] = (_Float16)bpre[0][c];
                h[1] = (_Float16)bpre[1][c];
                *(v2h*)&Bs[cur][bn + c][bk] = h;
            }
        }
        __syncthreads();
        wmma_step<C, MT, NT>(As[cur], Bs[cur], wm, wn, l15, half, acc);
    }

    // ---- epilogue: bias + ReLU + store (C/D layout: m = v + 8*half, n = l15)
#pragma unroll
    for (int mt = 0; mt < MT; ++mt) {
#pragma unroll
        for (int t = 0; t < NT; ++t) {
            const int n  = wn * (C / 4) + t * 16 + l15;
            const float bv = bias ? bias[n] : 0.0f;
#pragma unroll
            for (int v = 0; v < 8; ++v) {
                const int m = mBase + wm * 32 + mt * 16 + half * 8 + v;
                if (m < M) {
                    float r = acc[mt][t][v] + bv;
                    if (RELU) r = fmaxf(r, 0.0f);
                    D[(size_t)m * C + n] = r;
                }
            }
        }
    }
}

extern "C" void kernel_launch(void* const* d_in, const int* in_sizes, int n_in,
                              void* d_out, int out_size, void* d_ws, size_t ws_size,
                              hipStream_t stream) {
    (void)in_sizes; (void)n_in; (void)out_size; (void)ws_size;

    const float* x   = (const float*)d_in[0];  // [10000,512]
    const float* adj = (const float*)d_in[1];  // [10000,10000]
    const float* W1  = (const float*)d_in[2];  // [512,256]
    const float* b1  = (const float*)d_in[3];  // [256]
    const float* W2  = (const float*)d_in[4];  // [256,64]
    const float* b2  = (const float*)d_in[5];  // [64]
    float* out = (float*)d_out;                // [10000,64]

    const int N = 10000, IN = 512, HID = 256;

    // workspace: T1 | H | T2 (f32), ~23 MB total
    float* T1 = (float*)d_ws;                   // [N,HID]
    float* H  = T1 + (size_t)N * HID;           // [N,HID]
    float* T2 = H  + (size_t)N * HID;           // [N,64]

    dim3 blk(256);
    dim3 grd((N + MTILE - 1) / MTILE);          // 157 blocks

    gemm_wmma_f16<256, false><<<grd, blk, 0, stream>>>(x,   W1, nullptr, T1, N, IN);
    gemm_wmma_f16<256, true ><<<grd, blk, 0, stream>>>(adj, T1, b1,      H,  N, N);
    gemm_wmma_f16<64,  false><<<grd, blk, 0, stream>>>(H,   W2, nullptr, T2, N, HID);
    gemm_wmma_f16<64,  false><<<grd, blk, 0, stream>>>(adj, T2, b2,      out, N, N);
}
